// BahdanauAttention_46437186404897
// MI455X (gfx1250) — matrix operand order repaired and checked by reading, not compiled, not run on hardware
//
#include <hip/hip_runtime.h>
#include <hip/hip_bf16.h>

typedef __attribute__((ext_vector_type(16))) __bf16 v16bf;
typedef __attribute__((ext_vector_type(8)))  float  v8f;
typedef __attribute__((ext_vector_type(4)))  int    v4i;
typedef __attribute__((address_space(1))) v4i* g_v4i_p;   // global int4*
typedef __attribute__((address_space(3))) v4i* l_v4i_p;   // LDS int4*

#define NEG_INF_F (-1e9f)

#if __has_builtin(__builtin_amdgcn_global_load_async_to_lds_b128)
#define HAVE_ASYNC_LDS 1
#else
#define HAVE_ASYNC_LDS 0
#endif

__device__ __forceinline__ unsigned short f2bf(float f) {
  union { float f; unsigned u; } v; v.f = f;
  unsigned u = v.u;
  unsigned r = 0x7FFFu + ((u >> 16) & 1u);   // round-to-nearest-even
  return (unsigned short)((u + r) >> 16);
}

union BF16x16 { v16bf v; unsigned short u[16]; uint4 q[2]; };

__device__ __forceinline__ float fast_tanh(float x) {
#if __has_builtin(__builtin_amdgcn_tanhf)
  return __builtin_amdgcn_tanhf(x);           // V_TANH_F32 on gfx1250
#else
  float r;
  asm volatile("v_tanh_f32 %0, %1\n\tv_nop" : "=v"(r) : "v"(x));
  return r;
#endif
}

__device__ __forceinline__ void wait_async0() {
#if __has_builtin(__builtin_amdgcn_s_wait_asynccnt)
  __builtin_amdgcn_s_wait_asynccnt(0);
#else
  asm volatile("s_wait_asynccnt 0x0" ::: "memory");
#endif
}

#if HAVE_ASYNC_LDS
__device__ __forceinline__ void async_b128(const float* gsrc, float* ldst) {
  __builtin_amdgcn_global_load_async_to_lds_b128(
      (g_v4i_p)gsrc, (l_v4i_p)ldst, 0, 0);
}
#endif

// ---------------------------------------------------------------------------
// Transpose + convert: In (batched) [R,C] f32  ->  Out [C,R] bf16
// ---------------------------------------------------------------------------
__global__ void __launch_bounds__(256)
ktr_f32_to_bf16T(const float* __restrict__ In, unsigned short* __restrict__ Out,
                 int R, int C) {
  __shared__ unsigned short tile[32][33];
  const int bz = blockIdx.z;
  const float* in = In + (size_t)bz * R * C;
  unsigned short* out = Out + (size_t)bz * C * R;
  const int x = blockIdx.x * 32 + threadIdx.x;         // C index
  #pragma unroll
  for (int k = 0; k < 4; ++k) {
    int y = blockIdx.y * 32 + threadIdx.y + k * 8;     // R index
    tile[threadIdx.y + k * 8][threadIdx.x] = f2bf(in[(size_t)y * C + x]);
  }
  __syncthreads();
  const int xo = blockIdx.y * 32 + threadIdx.x;        // R index (inner out dim)
  #pragma unroll
  for (int k = 0; k < 4; ++k) {
    int yo = blockIdx.x * 32 + threadIdx.y + k * 8;    // C index (outer out dim)
    out[(size_t)yo * R + xo] = tile[threadIdx.x][threadIdx.y + k * 8];
  }
}

// ---------------------------------------------------------------------------
// Projection GEMM: Out[M,U] (f32) = A[M,D] (f32) @ W[D,U], W given as
// WT[U,D] bf16. One 16x16 tile per wave, bf16 WMMA with f32 accumulation.
// ---------------------------------------------------------------------------
__global__ void __launch_bounds__(256)
kp_proj(const float* __restrict__ A, const unsigned short* __restrict__ WT,
        float* __restrict__ Out, int M, int D, int U) {
  const int w    = threadIdx.x >> 5;
  const int lane = threadIdx.x & 31;
  const int h    = lane >> 4;          // half-wave
  const int n    = lane & 15;
  const int m0 = (blockIdx.x * 8 + w) * 16;
  const int n0 = blockIdx.y * 16;

  const float*          arow = A  + (size_t)(m0 + n) * D;   // A: M = lane%16
  const unsigned short* brow = WT + (size_t)(n0 + n) * D;   // B: N = lane%16
  v8f c = {};
  for (int kc = 0; kc < D; kc += 32) {
    // A operand: elems[0..7] = K(kc+8h..+7), elems[8..15] = K(kc+16+8h..+7)
    const float4* ap0 = (const float4*)(arow + kc + 8 * h);
    const float4* ap1 = (const float4*)(arow + kc + 16 + 8 * h);
    float4 a0 = ap0[0], a1 = ap0[1], a2 = ap1[0], a3 = ap1[1];
    BF16x16 av;
    av.u[0]  = f2bf(a0.x); av.u[1]  = f2bf(a0.y); av.u[2]  = f2bf(a0.z); av.u[3]  = f2bf(a0.w);
    av.u[4]  = f2bf(a1.x); av.u[5]  = f2bf(a1.y); av.u[6]  = f2bf(a1.z); av.u[7]  = f2bf(a1.w);
    av.u[8]  = f2bf(a2.x); av.u[9]  = f2bf(a2.y); av.u[10] = f2bf(a2.z); av.u[11] = f2bf(a2.w);
    av.u[12] = f2bf(a3.x); av.u[13] = f2bf(a3.y); av.u[14] = f2bf(a3.z); av.u[15] = f2bf(a3.w);
    // B operand: elems[0..7] = K(kc+8h..+7), elems[8..15] = K(kc+16+8h..+7)
    BF16x16 bv;
    bv.q[0] = *(const uint4*)(brow + kc + 8 * h);
    bv.q[1] = *(const uint4*)(brow + kc + 16 + 8 * h);
    c = __builtin_amdgcn_wmma_f32_16x16x32_bf16(false, av.v, false, bv.v,
                                                (short)0, c, false, false);
  }
  #pragma unroll
  for (int r = 0; r < 8; ++r)   // C/D: lane -> (M = r + 8h, N = lane%16)
    Out[(size_t)(m0 + 8 * h + r) * U + n0 + n] = c[r];
}

// ---------------------------------------------------------------------------
// Scores + softmax: per (b, 16-query tile) compute all Tv additive scores
//   s[i,j] = sum_u scale[u]*tanh(q[i,u]+k[j,u]),  mask -> -1e9, softmax,
// write weights as bf16 [B,Tq,Tv]. Key chunks staged via async LDS DMA.
// ---------------------------------------------------------------------------
__global__ void __launch_bounds__(256)
kscore(const float* __restrict__ Q, const float* __restrict__ Kp,
       const unsigned char* __restrict__ mask, const float* __restrict__ scale,
       unsigned short* __restrict__ Wout, int Tq, int Tv, int U) {
  __shared__ float qs[16][132];      // padded: conflict-free column reads
  __shared__ float ks[32][132];
  __shared__ float sc[16][516];      // stride 516 -> 4*i bank skew
  __shared__ float red[16][16];
  __shared__ float scl[128];
  __shared__ float rowm[16], rowl[16];

  const int t  = threadIdx.x;
  const int b  = blockIdx.y;
  const int i0 = blockIdx.x * 16;

  if (t < 128) scl[t] = scale[t];
#if HAVE_ASYNC_LDS
  #pragma unroll
  for (int r = 0; r < 2; ++r) {      // 512 float4s: memory -> LDS direct
    int f = t + r * 256;
    int i = f >> 5, u = (f & 31) * 4;
    async_b128(&Q[(size_t)(b * Tq + i0 + i) * U + u], &qs[i][u]);
  }
  wait_async0();
#else
  for (int idx = t; idx < 16 * 128; idx += 256) {
    int i = idx >> 7, u = idx & 127;
    qs[i][u] = Q[(size_t)(b * Tq + i0 + i) * U + u];
  }
#endif
  __syncthreads();

  const int jl = t & 31;             // key column within chunk
  const int iw = t >> 5;             // this thread handles rows iw and iw+8
  for (int kc = 0; kc < Tv; kc += 32) {
#if HAVE_ASYNC_LDS
    #pragma unroll
    for (int r = 0; r < 4; ++r) {    // 1024 float4s: memory -> LDS direct
      int f = t + r * 256;
      int j = f >> 5, u = (f & 31) * 4;
      async_b128(&Kp[(size_t)(b * Tv + kc + j) * U + u], &ks[j][u]);
    }
#else
    for (int idx = t; idx < 32 * 128; idx += 256) {
      int j = idx >> 7, u = idx & 127;
      ks[j][u] = Kp[(size_t)(b * Tv + kc + j) * U + u];
    }
#endif
    if (kc + 32 < Tv)                 // hint next chunk toward L2
      __builtin_prefetch(&Kp[(size_t)(b * Tv + kc + 32 + jl) * U + iw * 16], 0, 1);
#if HAVE_ASYNC_LDS
    wait_async0();
#endif
    __syncthreads();

    float acc0 = 0.f, acc1 = 0.f;
    const float* kr = ks[jl];
    const float* q0 = qs[iw];
    const float* q1 = qs[iw + 8];
    #pragma unroll 4
    for (int u = 0; u < 128; ++u) {
      float kv = kr[u], sv = scl[u];
      acc0 += sv * fast_tanh(q0[u] + kv);
      acc1 += sv * fast_tanh(q1[u] + kv);
    }
    bool ok = mask[(size_t)b * Tv + kc + jl] != 0;
    sc[iw][kc + jl]     = ok ? acc0 : NEG_INF_F;
    sc[iw + 8][kc + jl] = ok ? acc1 : NEG_INF_F;
    __syncthreads();
  }

  // row softmax: 16 threads per row, 32-wide chunks
  const int i = t & 15, c = t >> 4;
  float m = -3.4e38f;
  for (int j = c * 32; j < c * 32 + 32; ++j) m = fmaxf(m, sc[i][j]);
  red[i][c] = m;
  __syncthreads();
  if (t < 16) {
    float mm = red[t][0];
    for (int cc = 1; cc < 16; ++cc) mm = fmaxf(mm, red[t][cc]);
    rowm[t] = mm;
  }
  __syncthreads();
  const float mi = rowm[i];
  float sme = 0.f;
  for (int j = c * 32; j < c * 32 + 32; ++j) sme += __expf(sc[i][j] - mi);
  red[i][c] = sme;
  __syncthreads();
  if (t < 16) {
    float s = 0.f;
    for (int cc = 0; cc < 16; ++cc) s += red[t][cc];
    rowl[t] = s;
  }
  __syncthreads();
  for (int idx = t; idx < 16 * 512; idx += 256) {
    int ii = idx >> 9, jj = idx & 511;
    float wv = __expf(sc[ii][jj] - rowm[ii]) / rowl[ii];
    Wout[(size_t)(b * Tq + i0 + ii) * Tv + jj] = f2bf(wv);
  }
}

// ---------------------------------------------------------------------------
// Context GEMM: Out[b,Tq,D] f32 = Wgt[b,Tq,Tv] bf16 @ value (as VT[b,D,Tv] bf16)
// ---------------------------------------------------------------------------
__global__ void __launch_bounds__(256)
kctx(const unsigned short* __restrict__ Wgt, const unsigned short* __restrict__ VT,
     float* __restrict__ Out, int Tq, int Tv, int D) {
  const int w    = threadIdx.x >> 5;
  const int lane = threadIdx.x & 31;
  const int h    = lane >> 4;
  const int n    = lane & 15;
  const int b  = blockIdx.z;
  const int m0 = (blockIdx.x * 8 + w) * 16;
  const int n0 = blockIdx.y * 16;

  const unsigned short* arow = Wgt + (size_t)b * Tq * Tv + (size_t)(m0 + n) * Tv;
  const unsigned short* brow = VT  + (size_t)b * D  * Tv + (size_t)(n0 + n) * Tv;
  v8f c = {};
  for (int kc = 0; kc < Tv; kc += 32) {
    BF16x16 av, bv;
    av.q[0] = *(const uint4*)(arow + kc + 8 * h);
    av.q[1] = *(const uint4*)(arow + kc + 16 + 8 * h);
    bv.q[0] = *(const uint4*)(brow + kc + 8 * h);
    bv.q[1] = *(const uint4*)(brow + kc + 16 + 8 * h);
    c = __builtin_amdgcn_wmma_f32_16x16x32_bf16(false, av.v, false, bv.v,
                                                (short)0, c, false, false);
  }
  float* obase = Out + (size_t)b * Tq * D;
  #pragma unroll
  for (int r = 0; r < 8; ++r)
    obase[(size_t)(m0 + 8 * h + r) * D + n0 + n] = c[r];
}

// ---------------------------------------------------------------------------
extern "C" void kernel_launch(void* const* d_in, const int* in_sizes, int n_in,
                              void* d_out, int out_size, void* d_ws, size_t ws_size,
                              hipStream_t stream) {
  (void)in_sizes; (void)n_in; (void)out_size; (void)ws_size;
  const float*         query = (const float*)d_in[0];
  const float*         key   = (const float*)d_in[1];
  const float*         value = (const float*)d_in[2];
  const unsigned char* mask  = (const unsigned char*)d_in[3];
  const float*         Wa    = (const float*)d_in[4];
  const float*         Ua    = (const float*)d_in[5];
  const float*         scale = (const float*)d_in[6];

  const int B = 8, Tq = 512, Tv = 512, D = 512, U = 128;

  char* ws = (char*)d_ws;
  float*          Qw  = (float*)(ws);                                    // 2 MB
  float*          Kw  = (float*)(ws + (size_t)2 * 1024 * 1024);          // 2 MB
  unsigned short* WaT = (unsigned short*)(ws + (size_t)4 * 1024 * 1024);             // 128 KB
  unsigned short* UaT = (unsigned short*)(ws + (size_t)4 * 1024 * 1024 + 128 * 1024);// 128 KB
  unsigned short* VTb = (unsigned short*)(ws + (size_t)4 * 1024 * 1024 + 256 * 1024);// 4 MB
  unsigned short* Wgt = (unsigned short*)(ws + (size_t)8 * 1024 * 1024 + 256 * 1024);// 4 MB

  dim3 tb(32, 8);
  // Wa [D,U] -> WaT [U,D] bf16 ; Ua likewise
  ktr_f32_to_bf16T<<<dim3(U / 32, D / 32, 1), tb, 0, stream>>>(Wa, WaT, D, U);
  ktr_f32_to_bf16T<<<dim3(U / 32, D / 32, 1), tb, 0, stream>>>(Ua, UaT, D, U);
  // value [B,Tv,D] -> VT [B,D,Tv] bf16
  ktr_f32_to_bf16T<<<dim3(D / 32, Tv / 32, B), tb, 0, stream>>>(value, VTb, Tv, D);
  // projections
  kp_proj<<<dim3(B * Tq / 128, U / 16), 256, 0, stream>>>(query, WaT, Qw, B * Tq, D, U);
  kp_proj<<<dim3(B * Tv / 128, U / 16), 256, 0, stream>>>(key,   UaT, Kw, B * Tv, D, U);
  // additive scores + masked softmax -> bf16 weights
  kscore<<<dim3(Tq / 16, B), 256, 0, stream>>>(Qw, Kw, mask, scale, Wgt, Tq, Tv, U);
  // context = weights @ value
  kctx<<<dim3(Tq / 16 / 8, D / 16, B), 256, 0, stream>>>(Wgt, VTb, (float*)d_out, Tq, Tv, D);
}
